// ContinuousMamba_34866544509341
// MI455X (gfx1250) — compile-verified, matrix operand order checked by reading, not run on hardware
//
#include <hip/hip_runtime.h>
#include <hip/hip_bf16.h>

// ---------------- model constants ----------------
#define NB    4      // batch
#define LSEQ  1025   // NUM_X + SEQ_V
#define MREAL (NB*LSEQ)      // 4100 rows
#define MPAD  4128           // 129 * 32  (M-blocks of 32)
#define MBLK  129            // number of 32-row macro blocks
#define DM    768
#define DI    1536
#define DXZ   (2*DI)         // 3072
#define DST   16             // d_state
#define DTR   48             // dt_rank
#define DTRP  64             // padded dt_rank (multiple of 32)
#define XDBL  (DTR + 2*DST)  // 80
#define XDBLP 128            // N-padded x_proj rows (multiple of 64)

typedef __attribute__((ext_vector_type(16))) __bf16 v16bf;
typedef __attribute__((ext_vector_type(8)))  __bf16 v8bf;
typedef __attribute__((ext_vector_type(8)))  float  v8f;

__device__ __forceinline__ unsigned short f2bf(float f) {
    unsigned int u = __float_as_uint(f);
    unsigned int r = u + 0x7FFFu + ((u >> 16) & 1u);   // round-to-nearest-even
    return (unsigned short)(r >> 16);
}

__device__ __forceinline__ v16bf load_frag(const __bf16* __restrict__ row, int k0, int half) {
    // 16-bit A/B fragment: half 0 holds K {0..7,16..23}, half 1 holds {8..15,24..31}
    v8bf c0 = *reinterpret_cast<const v8bf*>(row + k0 + half * 8);
    v8bf c1 = *reinterpret_cast<const v8bf*>(row + k0 + 16 + half * 8);
    v16bf r;
#pragma unroll
    for (int i = 0; i < 8; ++i) { r[i] = c0[i]; r[i + 8] = c1[i]; }
    return r;
}

// ---------------- elementwise f32 -> bf16 ----------------
__global__ __launch_bounds__(256) void cvt_bf16_kernel(const float* __restrict__ src,
                                                       unsigned short* __restrict__ dst,
                                                       long n) {
    long i = (long)blockIdx.x * blockDim.x + threadIdx.x;
    if (i < n) dst[i] = f2bf(src[i]);
}

__global__ __launch_bounds__(256) void zero_u16_kernel(unsigned short* __restrict__ p, long n) {
    long i = (long)blockIdx.x * blockDim.x + threadIdx.x;
    if (i < n) p[i] = 0;
}

// dt_proj_w (L,1536,48) f32 -> (L,1536,64) bf16, K-padded with zeros
__global__ __launch_bounds__(256) void stage_dtw_kernel(const float* __restrict__ w,
                                                        unsigned short* __restrict__ out) {
    long i = (long)blockIdx.x * blockDim.x + threadIdx.x;   // over 4*1536*64
    if (i >= (long)4 * DI * DTRP) return;
    int kp = (int)(i % DTRP);
    long nl = i / DTRP;
    float v = (kp < DTR) ? w[nl * DTR + kp] : 0.f;
    out[i] = f2bf(v);
}

// x_proj_w (L,80,1536) f32 -> (L,128,1536) bf16, N-padded with zero rows
__global__ __launch_bounds__(256) void stage_xpw_kernel(const float* __restrict__ w,
                                                        unsigned short* __restrict__ out) {
    long i = (long)blockIdx.x * blockDim.x + threadIdx.x;   // over 4*128*1536
    if (i >= (long)4 * XDBLP * DI) return;
    int k = (int)(i % DI);
    long r = i / DI;
    int n = (int)(r % XDBLP);
    int l = (int)(r / XDBLP);
    float v = (n < XDBL) ? w[((long)l * XDBL + n) * DI + k] : 0.f;
    out[i] = f2bf(v);
}

// x_dbl (MPAD,80) f32 -> dt bf16 (MPAD,64), K-padded, pad rows zero
__global__ __launch_bounds__(256) void stage_dt_kernel(const float* __restrict__ xdbl,
                                                       unsigned short* __restrict__ dtb) {
    long i = (long)blockIdx.x * blockDim.x + threadIdx.x;   // over MPAD*64
    if (i >= (long)MPAD * DTRP) return;
    int kp = (int)(i % DTRP);
    long m = i / DTRP;
    float v = (kp < DTR && m < MREAL) ? xdbl[m * XDBL + kp] : 0.f;
    dtb[i] = f2bf(v);
}

// ---------------- embedding: h[b,t,:] f32 + bf16, pad rows zero ----------------
__global__ __launch_bounds__(256) void embed_kernel(const float* __restrict__ x_in,
                                                    const float* __restrict__ v_in,
                                                    const float* __restrict__ xw,
                                                    const float* __restrict__ xb,
                                                    const float* __restrict__ vw,
                                                    const float* __restrict__ vb,
                                                    float* __restrict__ hf,
                                                    unsigned short* __restrict__ hb) {
    long i = (long)blockIdx.x * blockDim.x + threadIdx.x;   // over MPAD*DM
    if (i >= (long)MPAD * DM) return;
    int d = (int)(i % DM);
    long m = i / DM;
    float val = 0.f;
    if (m < MREAL) {
        int b = (int)(m / LSEQ), t = (int)(m % LSEQ);
        if (t == 0) {
            val = x_in[b*2+0]*xw[d*2+0] + x_in[b*2+1]*xw[d*2+1] + xb[d];
        } else {
            long s = (long)b * 1024 + (t - 1);
            val = v_in[s*2+0]*vw[d*2+0] + v_in[s*2+1]*vw[d*2+1] + vb[d];
        }
    }
    hf[i] = val;
    hb[i] = f2bf(val);
}

// ---------------- WMMA GEMM: C[M,N] = A[M,K] @ W[N,K]^T ----------------
// Register-blocked: each wave owns a 32x64 macro-tile (2 M-tiles x 4 N-tiles,
// 8 accumulators). Per K-step: 2 A frags + 4 B frags -> 8 WMMAs.
// Nw = padded weight rows (mult of 64); Nout = real N = output stride.
// mode 0: plain; mode 1: +bias then softplus. Cbf optional bf16 mirror.
__global__ __launch_bounds__(256) void gemm_wmma_bf16_kernel(
        const unsigned short* __restrict__ Au, const unsigned short* __restrict__ Wu,
        const float* __restrict__ bias, float* __restrict__ Cf,
        unsigned short* __restrict__ Cbf, int Nw, int Nout, int K, int mode,
        int total_tiles) {
    int lane = threadIdx.x & 31;
    int wave = threadIdx.x >> 5;
    int tile = blockIdx.x * 8 + wave;
    if (tile >= total_tiles) return;
    int nBlkCount = Nw >> 6;
    int nb = tile % nBlkCount;          // 64-col block
    int mb = tile / nBlkCount;          // 32-row block
    int half = lane >> 4;
    int l16  = lane & 15;

    const __bf16* A = reinterpret_cast<const __bf16*>(Au);
    const __bf16* W = reinterpret_cast<const __bf16*>(Wu);
    const __bf16* Ar0 = A + (size_t)(mb * 32 + l16) * K;
    const __bf16* Ar1 = Ar0 + (size_t)16 * K;
    const __bf16* Wr0 = W + (size_t)(nb * 64 + l16) * K;

    v8f acc[2][4] = {};
#pragma unroll 1
    for (int k0 = 0; k0 < K; k0 += 32) {
        v16bf a0 = load_frag(Ar0, k0, half);
        v16bf a1 = load_frag(Ar1, k0, half);
        v16bf b0 = load_frag(Wr0,                    k0, half);
        v16bf b1 = load_frag(Wr0 + (size_t)16 * K,   k0, half);
        v16bf b2 = load_frag(Wr0 + (size_t)32 * K,   k0, half);
        v16bf b3 = load_frag(Wr0 + (size_t)48 * K,   k0, half);
        acc[0][0] = __builtin_amdgcn_wmma_f32_16x16x32_bf16(false, a0, false, b0, (short)0, acc[0][0], false, false);
        acc[0][1] = __builtin_amdgcn_wmma_f32_16x16x32_bf16(false, a0, false, b1, (short)0, acc[0][1], false, false);
        acc[0][2] = __builtin_amdgcn_wmma_f32_16x16x32_bf16(false, a0, false, b2, (short)0, acc[0][2], false, false);
        acc[0][3] = __builtin_amdgcn_wmma_f32_16x16x32_bf16(false, a0, false, b3, (short)0, acc[0][3], false, false);
        acc[1][0] = __builtin_amdgcn_wmma_f32_16x16x32_bf16(false, a1, false, b0, (short)0, acc[1][0], false, false);
        acc[1][1] = __builtin_amdgcn_wmma_f32_16x16x32_bf16(false, a1, false, b1, (short)0, acc[1][1], false, false);
        acc[1][2] = __builtin_amdgcn_wmma_f32_16x16x32_bf16(false, a1, false, b2, (short)0, acc[1][2], false, false);
        acc[1][3] = __builtin_amdgcn_wmma_f32_16x16x32_bf16(false, a1, false, b3, (short)0, acc[1][3], false, false);
    }

#pragma unroll
    for (int ni = 0; ni < 4; ++ni) {
        int col = nb * 64 + ni * 16 + l16;
        if (col >= Nout) continue;
        float bv = (bias != nullptr) ? bias[col] : 0.f;
#pragma unroll
        for (int mi = 0; mi < 2; ++mi) {
#pragma unroll
            for (int v = 0; v < 8; ++v) {
                int row = mb * 32 + mi * 16 + v + 8 * half;
                float val = acc[mi][ni][v] + bv;
                if (mode == 1) {                       // softplus (stable)
                    val = (val > 20.f) ? val : log1pf(__expf(val));
                }
                size_t o = (size_t)row * Nout + col;
                Cf[o] = val;
                if (Cbf) Cbf[o] = f2bf(val);
            }
        }
    }
}

// ---------------- causal depthwise conv (k=4) + bias + SiLU ----------------
__global__ __launch_bounds__(256) void conv_silu_kernel(const float* __restrict__ xz,
                                                        const float* __restrict__ cw,
                                                        const float* __restrict__ cb,
                                                        float* __restrict__ xc,
                                                        unsigned short* __restrict__ xcb) {
    long i = (long)blockIdx.x * blockDim.x + threadIdx.x;   // over MPAD*DI
    if (i >= (long)MPAD * DI) return;
    int d = (int)(i % DI);
    long m = i / DI;
    float val = 0.f;
    if (m < MREAL) {
        int b = (int)(m / LSEQ), t = (int)(m % LSEQ);
        float acc = cb[d];
#pragma unroll
        for (int j = 0; j < 4; ++j) {
            int ts = t - 3 + j;
            if (ts >= 0)
                acc += cw[d * 4 + j] * xz[(size_t)(b * LSEQ + ts) * DXZ + d];
        }
        val = acc / (1.f + __expf(-acc));   // SiLU
    }
    xc[i] = val;
    xcb[i] = f2bf(val);
}

// ---------------- selective scan (serial over t) + D-skip + SiLU(z) gate ----------------
// one thread per (batch, channel); B/C broadcast via LDS per step
__global__ __launch_bounds__(256) void scan_kernel(const float* __restrict__ xz,
                                                   const float* __restrict__ xconv,
                                                   const float* __restrict__ delta,
                                                   const float* __restrict__ xdbl,
                                                   const float* __restrict__ Alog,
                                                   const float* __restrict__ Dskip,
                                                   unsigned short* __restrict__ yb) {
    __shared__ float sBC[2 * DST];
    int b = blockIdx.x / 6;
    int d = (blockIdx.x % 6) * 256 + threadIdx.x;   // 0..1535

    float a[DST], hst[DST];
#pragma unroll
    for (int n = 0; n < DST; ++n) {
        a[n] = -__expf(Alog[d * DST + n]);
        hst[n] = 0.f;
    }
    float Dd = Dskip[d];

    for (int t = 0; t < LSEQ; ++t) {
        size_t m = (size_t)b * LSEQ + t;
        __syncthreads();
        if (threadIdx.x < 2 * DST)
            sBC[threadIdx.x] = xdbl[m * XDBL + DTR + threadIdx.x];
        __syncthreads();

        float u  = xconv[m * DI + d];
        float dl = delta[m * DI + d];
        float du = dl * u;
        float y  = 0.f;
#pragma unroll
        for (int n = 0; n < DST; ++n) {
            float dA = __expf(dl * a[n]);
            hst[n] = dA * hst[n] + du * sBC[n];
            y += hst[n] * sBC[DST + n];
        }
        y += u * Dd;
        float z  = xz[m * DXZ + DI + d];
        float sz = z / (1.f + __expf(-z));
        yb[m * DI + d] = f2bf(y * sz);
    }
}

// ---------------- head: y_preds[b,s,yd] (skips t==0) ----------------
__global__ __launch_bounds__(256) void head_kernel(const float* __restrict__ hf,
                                                   const float* __restrict__ hw,
                                                   const float* __restrict__ hbias,
                                                   float* __restrict__ out) {
    int id = blockIdx.x * blockDim.x + threadIdx.x;   // 4*1024*2
    if (id >= NB * 1024 * 2) return;
    int yd = id & 1;
    int s  = (id >> 1) & 1023;
    int b  = id >> 11;
    const float* hrow = hf + (size_t)(b * LSEQ + s + 1) * DM;
    float acc = hbias[yd];
    for (int k = 0; k < DM; ++k) acc += hrow[k] * hw[yd * DM + k];
    out[id] = acc;
}

// ---------------- launcher ----------------
extern "C" void kernel_launch(void* const* d_in, const int* in_sizes, int n_in,
                              void* d_out, int out_size, void* d_ws, size_t ws_size,
                              hipStream_t stream) {
    const float* x_inputs  = (const float*)d_in[0];
    const float* v_inputs  = (const float*)d_in[1];
    const float* x_emb_w   = (const float*)d_in[2];
    const float* x_emb_b   = (const float*)d_in[3];
    const float* v_proj_w  = (const float*)d_in[4];
    const float* v_proj_b  = (const float*)d_in[5];
    const float* head_w    = (const float*)d_in[6];
    const float* head_b    = (const float*)d_in[7];
    const float* in_proj_w = (const float*)d_in[8];
    const float* conv_w    = (const float*)d_in[9];
    const float* conv_b    = (const float*)d_in[10];
    const float* x_proj_w  = (const float*)d_in[11];
    const float* dt_proj_w = (const float*)d_in[12];
    const float* dt_proj_b = (const float*)d_in[13];
    const float* A_log     = (const float*)d_in[14];
    const float* D_skip    = (const float*)d_in[15];
    const float* out_proj_w= (const float*)d_in[16];
    float* out = (float*)d_out;

    size_t off = 0;
    auto alloc = [&](size_t bytes) -> void* {
        void* p = (char*)d_ws + off;
        off += (bytes + 255) & ~(size_t)255;
        return p;
    };
    float*          hf    = (float*)         alloc((size_t)MPAD * DM  * 4);
    unsigned short* hb    = (unsigned short*)alloc((size_t)MPAD * DM  * 2);
    float*          xz    = (float*)         alloc((size_t)MPAD * DXZ * 4);
    float*          xconv = (float*)         alloc((size_t)MPAD * DI  * 4);
    unsigned short* xcb   = (unsigned short*)alloc((size_t)MPAD * DI  * 2);
    float*          xdbl  = (float*)         alloc((size_t)MPAD * XDBL* 4);
    unsigned short* dtb   = (unsigned short*)alloc((size_t)MPAD * DTRP* 2);
    float*          dlt   = (float*)         alloc((size_t)MPAD * DI  * 4);
    unsigned short* yb    = (unsigned short*)alloc((size_t)MPAD * DI  * 2);
    unsigned short* w_in  = (unsigned short*)alloc((size_t)4 * DXZ * DM   * 2);
    unsigned short* w_out = (unsigned short*)alloc((size_t)4 * DM * DI    * 2);
    unsigned short* w_xp  = (unsigned short*)alloc((size_t)4 * XDBLP * DI * 2);
    unsigned short* w_dt  = (unsigned short*)alloc((size_t)4 * DI * DTRP  * 2);

    auto cdiv = [](long a, long b) { return (a + b - 1) / b; };

    // weight staging (bf16)
    long nInW  = (long)4 * DXZ * DM;
    long nOutW = (long)4 * DM * DI;
    cvt_bf16_kernel<<<cdiv(nInW, 256),  256, 0, stream>>>(in_proj_w,  w_in,  nInW);
    cvt_bf16_kernel<<<cdiv(nOutW, 256), 256, 0, stream>>>(out_proj_w, w_out, nOutW);
    stage_xpw_kernel<<<cdiv((long)4 * XDBLP * DI, 256), 256, 0, stream>>>(x_proj_w, w_xp);
    stage_dtw_kernel<<<cdiv((long)4 * DI * DTRP, 256), 256, 0, stream>>>(dt_proj_w, w_dt);

    // zero pad rows of yb (scan never writes them; out_proj reads them)
    zero_u16_kernel<<<cdiv((long)(MPAD - MREAL) * DI, 256), 256, 0, stream>>>(
        yb + (size_t)MREAL * DI, (long)(MPAD - MREAL) * DI);

    // embedding
    embed_kernel<<<cdiv((long)MPAD * DM, 256), 256, 0, stream>>>(
        x_inputs, v_inputs, x_emb_w, x_emb_b, v_proj_w, v_proj_b, hf, hb);

    for (int l = 0; l < 4; ++l) {
        // in_proj: (MPAD,768) @ (3072,768)^T -> xz
        {
            int total = (DXZ / 64) * MBLK;   // 48*129
            gemm_wmma_bf16_kernel<<<cdiv(total, 8), 256, 0, stream>>>(
                hb, w_in + (size_t)l * DXZ * DM, nullptr, xz, nullptr,
                DXZ, DXZ, DM, 0, total);
        }
        // depthwise causal conv + SiLU
        conv_silu_kernel<<<cdiv((long)MPAD * DI, 256), 256, 0, stream>>>(
            xz, conv_w + (size_t)l * DI * 4, conv_b + (size_t)l * DI, xconv, xcb);
        // x_proj: (MPAD,1536) @ (80,1536)^T -> x_dbl (weights N-padded to 128)
        {
            int total = (XDBLP / 64) * MBLK; // 2*129
            gemm_wmma_bf16_kernel<<<cdiv(total, 8), 256, 0, stream>>>(
                xcb, w_xp + (size_t)l * XDBLP * DI, nullptr, xdbl, nullptr,
                XDBLP, XDBL, DI, 0, total);
        }
        // stage dt (K-padded 48->64)
        stage_dt_kernel<<<cdiv((long)MPAD * DTRP, 256), 256, 0, stream>>>(xdbl, dtb);
        // dt_proj + bias + softplus -> delta
        {
            int total = (DI / 64) * MBLK;    // 24*129
            gemm_wmma_bf16_kernel<<<cdiv(total, 8), 256, 0, stream>>>(
                dtb, w_dt + (size_t)l * DI * DTRP, dt_proj_b + (size_t)l * DI,
                dlt, nullptr, DI, DI, DTRP, 1, total);
        }
        // selective scan + D-skip + SiLU(z) gate -> yb (bf16)
        scan_kernel<<<dim3(NB * 6), 256, 0, stream>>>(
            xz, xconv, dlt, xdbl, A_log + (size_t)l * DI * DST,
            D_skip + (size_t)l * DI, yb);
        // out_proj: (MPAD,1536) @ (768,1536)^T -> hf (f32) + hb (bf16)
        {
            int total = (DM / 64) * MBLK;    // 12*129
            gemm_wmma_bf16_kernel<<<cdiv(total, 8), 256, 0, stream>>>(
                yb, w_out + (size_t)l * DM * DI, nullptr, hf, hb,
                DM, DM, DI, 0, total);
        }
    }

    // head
    head_kernel<<<cdiv((long)NB * 1024 * 2, 256), 256, 0, stream>>>(
        hf, head_w, head_b, out);
}
